// ImprovedPolicyNetwork_22574348107935
// MI455X (gfx1250) — matrix operand order repaired and checked by reading, not compiled, not run on hardware
//
#include <hip/hip_runtime.h>
#include <hip/hip_bf16.h>

// =====================================================================
// ImprovedPolicyNetwork forward for MI455X (gfx1250, wave32, WMMA)
//
// All heavy math runs through v_wmma_f32_16x16x32_bf16 (bf16 operands,
// fp32 accumulate). Weights are converted fp32->bf16 into d_ws once per
// call (amortized over 256 recurrent steps). Weight tiles are staged
// into LDS with double-buffered GLOBAL_LOAD_ASYNC_TO_LDS_B128 DMA
// (ASYNCcnt-pipelined; falls back to VALU staging if the builtin is
// unavailable). Everything runs on `stream`, no host sync -> graph-safe.
//
// Workspace requirement: ~650 MB (combined[B,S,768] + 3x [B,S,512] fp32
// activations + bf16 weights + per-step scratch).
// =====================================================================

#define Bb   256
#define Ss   256
#define DIN  16
#define Hh   512
#define H2   256      // H/2
#define H4   2048     // 4H
#define CDd  768      // H + H/2
#define DXd  1280     // CD + H (decoder0 input)
#define NEGC (-1e9f)

typedef __attribute__((ext_vector_type(16))) __bf16 bf16x16;
typedef __attribute__((ext_vector_type(8)))  __bf16 bf16x8;
typedef __attribute__((ext_vector_type(8)))  float  f32x8;

__device__ __forceinline__ unsigned short f32_to_bf16_bits(float f) {
    unsigned int u = __float_as_uint(f);
    unsigned int lsb = (u >> 16) & 1u;
    u += 0x7FFFu + lsb;               // round-to-nearest-even
    return (unsigned short)(u >> 16);
}
__device__ __forceinline__ float sigf(float x) { return 1.0f / (1.0f + expf(-x)); }
__device__ __forceinline__ float clip9(float v) { return fminf(fmaxf(v, NEGC), -NEGC); }

// ---- CDNA5 async global->LDS DMA (guarded: falls back to VALU copy) ----
#if defined(__gfx1250__) && __has_builtin(__builtin_amdgcn_global_load_async_to_lds_b128)
#define HAVE_ASYNC_LDS 1
typedef __attribute__((ext_vector_type(4))) int i32x4;
typedef __attribute__((address_space(1))) i32x4 gvec128_t;   // global v4i
typedef __attribute__((address_space(3))) i32x4 lvec128_t;   // LDS v4i
__device__ __forceinline__ void async_ld_b128(const void* g, void* l) {
    // integer cast strips const and reinterprets into the target address
    // space (low 32 bits of a generic LDS pointer are the LDS offset per
    // the flat-aperture mapping).
    __builtin_amdgcn_global_load_async_to_lds_b128(
        (gvec128_t*)(unsigned long long)g,
        (lvec128_t*)(unsigned int)(unsigned long long)l, 0, 0);
}
#if __has_builtin(__builtin_amdgcn_s_wait_asynccnt)
#define WAIT_ASYNC(n) __builtin_amdgcn_s_wait_asynccnt(n)
#else
#define WAIT_ASYNC(n) asm volatile("s_wait_asynccnt " #n)
#endif
#endif

// ---------------------------------------------------------------------
// Generic bf16 WMMA GEMM:  C[M,N] = act( A[M,K](fp32) * W[N,K]^T + bias )
// W is bf16 bits, row-major [N,K]  (i.e. C = A @ W^T, cublas-TN style,
// which matches x @ wih.T directly).
// Tile: 64x64 per block of 128 threads (4 waves); K chunked by 32.
// W tiles double-buffered via async LDS DMA; A tiles VALU-converted.
// ---------------------------------------------------------------------
#define TM 64
#define TN 64
#define TK 32
#define LDT 40   // LDS row stride in shorts (80B, 16B aligned, pads banks)

enum { ACT_NONE = 0, ACT_RELU = 1 };

__global__ __launch_bounds__(128)
void k_gemm_bf16(const float* __restrict__ A, long lda,
                 const unsigned short* __restrict__ W,
                 const float* __restrict__ bias,
                 float* __restrict__ C, long ldc,
                 int M, int N, int K, int act)
{
    __shared__ __attribute__((aligned(16))) unsigned short sA[TM][LDT];
    __shared__ __attribute__((aligned(16))) unsigned short sB[2][TN][LDT];

    const int tid  = threadIdx.x;
    const int bm   = blockIdx.y * TM;
    const int bn   = blockIdx.x * TN;
    const int lane = tid & 31;
    const int wave = tid >> 5;
    const int wm   = (wave >> 1) * 32;   // wave row offset in tile
    const int wn   = (wave & 1) * 32;    // wave col offset in tile
    const int half = lane >> 4;          // 0: lanes 0-15, 1: lanes 16-31
    const int lr   = lane & 15;
    const int nt   = (K + TK - 1) / TK;

#ifdef HAVE_ASYNC_LDS
    // async path requires fully in-bounds W tiles (zero-fill not needed):
    // N edges fall back (only the tiny N=4 head), K edges fall back (K=16).
    const bool use_async = (bn + TN <= N) && ((K & (TK - 1)) == 0);
    if (use_async) {
        // prologue: DMA tile 0 into buffer 0 (2 b128 issues per lane;
        // 128 lanes x 2 x 16B = 4KB tile)
        #pragma unroll
        for (int q = 0; q < 2; ++q) {
            int chunk = q * 128 + tid;
            int row = chunk >> 2, ko = (chunk & 3) * 8;
            async_ld_b128(W + (long)(bn + row) * K + ko, &sB[0][row][ko]);
        }
    }
#else
    const bool use_async = false;
#endif

    f32x8 acc[2][2] = {};

    for (int it = 0; it < nt; ++it) {
        const int k0 = it * TK;
        const int cur = it & 1;
        // ---- prefetch next A K-slice (emits global_prefetch_b8) ----
        if (k0 + TK < K) {
            int pr = tid >> 1;  // 0..63
            if (bm + pr < M) __builtin_prefetch(&A[(long)(bm + pr) * lda + k0 + TK], 0, 1);
        }
        // ---- stage A (fp32 -> bf16, VALU) ----
        for (int idx = tid; idx < TM * TK; idx += 128) {
            int i = idx >> 5, j = idx & 31;
            int gm = bm + i, gk = k0 + j;
            float v = (gm < M && gk < K) ? A[(long)gm * lda + gk] : 0.0f;
            sA[i][j] = f32_to_bf16_bits(v);
        }
        // ---- stage W ----
        if (use_async) {
#ifdef HAVE_ASYNC_LDS
            if (it + 1 < nt) {
                // pipeline: DMA next tile into the other buffer, then wait
                // only for the current tile (in-order async completion).
                #pragma unroll
                for (int q = 0; q < 2; ++q) {
                    int chunk = q * 128 + tid;
                    int row = chunk >> 2, ko = (chunk & 3) * 8;
                    async_ld_b128(W + (long)(bn + row) * K + k0 + TK + ko,
                                  &sB[(it + 1) & 1][row][ko]);
                }
                WAIT_ASYNC(2);
            } else {
                WAIT_ASYNC(0);
            }
#endif
        } else {
            for (int idx = tid; idx < TN * TK; idx += 128) {
                int i = idx >> 5, j = idx & 31;
                int gn = bn + i, gk = k0 + j;
                sB[cur][i][j] = (gn < N && gk < K) ? W[(long)gn * K + gk] : (unsigned short)0;
            }
        }
        __syncthreads();

        // ---- build fragments per documented 16-bit VGPR layouts ----
        // A 16x32: lane half h, row lr: VGPR0-3 = K[h*8 .. h*8+7],
        //                               VGPR4-7 = K[16+h*8 .. 16+h*8+7]
        // B 32x16: lane half h, col lr: same K order as A
        bf16x16 afrag[2], bfrag[2];
        #pragma unroll
        for (int i = 0; i < 2; ++i) {
            const unsigned short* ap = &sA[wm + i * 16 + lr][half * 8];
            bf16x8 alo = *(const bf16x8*)ap;
            bf16x8 ahi = *(const bf16x8*)(ap + 16);
            #pragma unroll
            for (int e = 0; e < 8; ++e) { afrag[i][e] = alo[e]; afrag[i][8 + e] = ahi[e]; }

            const unsigned short* bp = &sB[cur][wn + i * 16 + lr][half * 8];
            bf16x8 b0 = *(const bf16x8*)bp;
            bf16x8 b1 = *(const bf16x8*)(bp + 16);
            #pragma unroll
            for (int e = 0; e < 8; ++e) { bfrag[i][e] = b0[e]; bfrag[i][8 + e] = b1[e]; }
        }
        #pragma unroll
        for (int i = 0; i < 2; ++i)
            #pragma unroll
            for (int j = 0; j < 2; ++j)
                acc[i][j] = __builtin_amdgcn_wmma_f32_16x16x32_bf16(
                    false, afrag[i], false, bfrag[j], (short)0, acc[i][j], false, false);
        __syncthreads();
    }

    // ---- store per documented C/D layout: VGPR r -> M = r + half*8 ----
    #pragma unroll
    for (int i = 0; i < 2; ++i) {
        #pragma unroll
        for (int j = 0; j < 2; ++j) {
            int col = bn + wn + j * 16 + lr;
            if (col >= N) continue;
            float bv = bias ? bias[col] : 0.0f;
            #pragma unroll
            for (int r = 0; r < 8; ++r) {
                int row = bm + wm + i * 16 + half * 8 + r;
                if (row < M) {
                    float v = acc[i][j][r] + bv;
                    if (act == ACT_RELU) v = fmaxf(v, 0.0f);
                    C[(long)row * ldc + col] = v;
                }
            }
        }
    }
}

// ---------------------------------------------------------------------
// Small utility / pointwise kernels
// ---------------------------------------------------------------------
__global__ void k_cast_bf16(const float* __restrict__ s, unsigned short* __restrict__ d, long n) {
    long i = (long)blockIdx.x * blockDim.x + threadIdx.x;
    if (i < n) d[i] = f32_to_bf16_bits(s[i]);
}
// src [R,C] fp32 row-major  ->  dst [C,R] bf16 (i.e. dst = src^T)
__global__ void k_castT_bf16(const float* __restrict__ s, unsigned short* __restrict__ d, int R, int C) {
    long i = (long)blockIdx.x * blockDim.x + threadIdx.x;
    if (i < (long)R * C) {
        int r = (int)(i / C), c = (int)(i % C);
        d[(long)c * R + r] = f32_to_bf16_bits(s[i]);
    }
}
__global__ void k_vadd(const float* a, const float* b, float* d, long n) {
    long i = (long)blockIdx.x * blockDim.x + threadIdx.x;
    if (i < n) d[i] = a[i] + b[i];
}
__global__ void k_zero(float* p, long n) {
    long i = (long)blockIdx.x * blockDim.x + threadIdx.x;
    if (i < n) p[i] = 0.0f;
}
// bin layer 1:  relu(bin_info[B,2] @ bin_w1[2,256] + bin_b1)
__global__ void k_bin1(const float* __restrict__ bi, const float* __restrict__ w1,
                       const float* __restrict__ b1, float* __restrict__ o) {
    int i = blockIdx.x * blockDim.x + threadIdx.x;
    if (i >= Bb * H2) return;
    int b = i / H2, j = i % H2;
    float v = bi[b * 2 + 0] * w1[j] + bi[b * 2 + 1] * w1[H2 + j] + b1[j];
    o[i] = fmaxf(v, 0.0f);
}
// combined[B,S,768] = [parts2 + pos_emb | broadcast(bin_enc)]
__global__ void k_combined(const float* __restrict__ parts2, const float* __restrict__ pos,
                           const float* __restrict__ binenc, float* __restrict__ comb) {
    long i = (long)blockIdx.x * blockDim.x + threadIdx.x;
    if (i >= (long)Bb * Ss * CDd) return;
    int c = (int)(i % CDd);
    long bs = i / CDd;
    int s = (int)(bs % Ss);
    int b = (int)(bs / Ss);
    float v;
    if (c < Hh) v = parts2[bs * Hh + c] + pos[(long)s * Hh + c];
    else        v = binenc[(long)b * H2 + (c - Hh)];
    comb[i] = v;
}
// LSTM cell: gates = gX + gH (biases folded into gX); PyTorch i,f,g,o order
__global__ void k_lstm_cell(const float* __restrict__ gX, const float* __restrict__ gH,
                            float* __restrict__ h, float* __restrict__ c,
                            float* __restrict__ hout, long hout_ld) {
    int i = blockIdx.x * blockDim.x + threadIdx.x;
    if (i >= Bb * Hh) return;
    int b = i / Hh, j = i % Hh;
    const float* gx = gX + (long)b * H4;
    const float* gh = gH + (long)b * H4;
    float ig = sigf(gx[j] + gh[j]);
    float fg = sigf(gx[Hh + j] + gh[Hh + j]);
    float gg = tanhf(gx[2 * Hh + j] + gh[2 * Hh + j]);
    float og = sigf(gx[3 * Hh + j] + gh[3 * Hh + j]);
    float cn = fg * c[i] + ig * gg;
    float hn = og * tanhf(cn);
    c[i] = cn; h[i] = hn;
    if (hout) hout[(long)b * hout_ld + j] = hn;
}
__global__ __launch_bounds__(256)
void k_layernorm(const float* __restrict__ x, float* __restrict__ y,
                 const float* __restrict__ g, const float* __restrict__ bt) {
    __shared__ float red[256];
    long row = blockIdx.x;
    const float* xr = x + row * Hh;
    float s = 0.0f;
    for (int j = threadIdx.x; j < Hh; j += 256) s += xr[j];
    red[threadIdx.x] = s; __syncthreads();
    for (int o = 128; o > 0; o >>= 1) { if (threadIdx.x < o) red[threadIdx.x] += red[threadIdx.x + o]; __syncthreads(); }
    float mu = red[0] / Hh; __syncthreads();
    float v = 0.0f;
    for (int j = threadIdx.x; j < Hh; j += 256) { float d = xr[j] - mu; v += d * d; }
    red[threadIdx.x] = v; __syncthreads();
    for (int o = 128; o > 0; o >>= 1) { if (threadIdx.x < o) red[threadIdx.x] += red[threadIdx.x + o]; __syncthreads(); }
    float rstd = rsqrtf(red[0] / Hh + 1e-5f); __syncthreads();
    for (int j = threadIdx.x; j < Hh; j += 256)
        y[row * Hh + j] = (xr[j] - mu) * rstd * g[j] + bt[j];
}
__global__ void k_mean_seq(const float* __restrict__ enc, float* __restrict__ m) {
    int i = blockIdx.x * blockDim.x + threadIdx.x;
    if (i >= Bb * Hh) return;
    int b = i / Hh, j = i % Hh;
    float s = 0.0f;
    for (int t = 0; t < Ss; ++t) s += enc[((long)b * Ss + t) * Hh + j];
    m[i] = s * (1.0f / Ss);
}
// decoder input gather (teacher forcing, ratio 1.0)
__global__ void k_dec_gather(const float* __restrict__ comb, const float* __restrict__ emean,
                             const int* __restrict__ tgt, float* __restrict__ dx, int t) {
    long i = (long)blockIdx.x * blockDim.x + threadIdx.x;
    if (i >= (long)Bb * DXd) return;
    int b = (int)(i / DXd), j = (int)(i % DXd);
    float v;
    if (t == 0) v = 0.0f;
    else if (j < CDd) {
        int tg = tgt[(long)b * Ss + (t - 1)];
        tg = min(max(tg, 0), Ss - 1);
        v = comb[((long)b * Ss + tg) * CDd + j];
    } else v = emean[(long)b * Hh + (j - CDd)];
    dx[i] = v;
}
// fused attention step: scores(tanh)->softmax->context; builds cc=[do|context]
__global__ __launch_bounds__(256)
void k_attn(const float* __restrict__ q, const float* __restrict__ eproj,
            const float* __restrict__ enc, const float* __restrict__ av,
            const float* __restrict__ dob, float* __restrict__ cc) {
    __shared__ float sq[Hh];
    __shared__ float sw[Ss];
    __shared__ float red[256];
    int b = blockIdx.x;
    for (int j = threadIdx.x; j < Hh; j += 256) sq[j] = q[(long)b * Hh + j];
    __syncthreads();
    int s = threadIdx.x;
    const float* ep = eproj + ((long)b * Ss + s) * Hh;
    float sc = 0.0f;
    for (int h = 0; h < Hh; ++h) sc += tanhf(sq[h] + ep[h]) * av[h];
    red[s] = sc; __syncthreads();
    for (int o = 128; o > 0; o >>= 1) { if (s < o) red[s] = fmaxf(red[s], red[s + o]); __syncthreads(); }
    float mx = red[0]; __syncthreads();
    float e = expf(sc - mx);
    red[s] = e; __syncthreads();
    for (int o = 128; o > 0; o >>= 1) { if (s < o) red[s] += red[s + o]; __syncthreads(); }
    float inv = 1.0f / red[0]; __syncthreads();
    sw[s] = e * inv; __syncthreads();
    for (int h = threadIdx.x; h < Hh; h += 256) {
        float ctx = 0.0f;
        for (int ss = 0; ss < Ss; ++ss) ctx += sw[ss] * enc[((long)b * Ss + ss) * Hh + h];
        cc[(long)b * 2 * Hh + Hh + h] = ctx;
        cc[(long)b * 2 * Hh + h]      = dob[(long)b * Hh + h];
    }
}
// logits: mask + clip + write, then update mask with this step's target
__global__ __launch_bounds__(256)
void k_logits(const float* __restrict__ lraw, float* __restrict__ mask,
              const int* __restrict__ tgt, float* __restrict__ out, int t) {
    int b = blockIdx.x, s = threadIdx.x;
    float v = clip9(lraw[(long)b * Ss + s] + mask[(long)b * Ss + s] * NEGC);
    out[((long)b * Ss + t) * Ss + s] = v;
    __syncthreads();
    if (s == 0) mask[(long)b * Ss + tgt[(long)b * Ss + t]] = 1.0f;
}
__global__ void k_rot(const float* __restrict__ rraw, float* __restrict__ outrot, int t) {
    int i = blockIdx.x * blockDim.x + threadIdx.x;
    if (i >= Bb * 4) return;
    int b = i >> 2, j = i & 3;
    outrot[((long)b * Ss + t) * 4 + j] = clip9(rraw[(long)b * 4 + j]);
}

// ---------------------------------------------------------------------
// Host-side launch orchestration
// ---------------------------------------------------------------------
static inline void gemm(hipStream_t st, const float* A, long lda, const unsigned short* W,
                        const float* bias, float* C, long ldc, int M, int N, int K, int act) {
    dim3 g((N + TN - 1) / TN, (M + TM - 1) / TM), b(128);
    hipLaunchKernelGGL(k_gemm_bf16, g, b, 0, st, A, lda, W, bias, C, ldc, M, N, K, act);
}
static inline void castw(hipStream_t st, const float* s, unsigned short* d, long n) {
    hipLaunchKernelGGL(k_cast_bf16, dim3((unsigned)((n + 255) / 256)), dim3(256), 0, st, s, d, n);
}
static inline void castT(hipStream_t st, const float* s, unsigned short* d, int R, int C) {
    long n = (long)R * C;
    hipLaunchKernelGGL(k_castT_bf16, dim3((unsigned)((n + 255) / 256)), dim3(256), 0, st, s, d, R, C);
}
static inline void zero(hipStream_t st, float* p, long n) {
    hipLaunchKernelGGL(k_zero, dim3((unsigned)((n + 255) / 256)), dim3(256), 0, st, p, n);
}

// input indices (setup_inputs dict order, params flattened in insertion order)
enum {
    IN_INPUTS = 0, IN_BIN, IN_TGT,
    P_BIN_W1, P_BIN_B1, P_BIN_W2, P_BIN_B2,
    P_PART_W1, P_PART_B1, P_PART_W2, P_PART_B2,
    P_POS,
    P_E0_WIH, P_E0_WHH, P_E0_BIH, P_E0_BHH,
    P_E1_WIH, P_E1_WHH, P_E1_BIH, P_E1_BHH,
    P_D0_WIH, P_D0_WHH, P_D0_BIH, P_D0_BHH,
    P_D1_WIH, P_D1_WHH, P_D1_BIH, P_D1_BHH,
    P_ATT_W, P_ATT_B, P_ATT_V,
    P_DW1, P_DB1, P_DW2, P_DB2,
    P_RW1, P_RB1, P_RW2, P_RB2,
    P_LN1_G, P_LN1_B, P_LN2_G, P_LN2_B
};

extern "C" void kernel_launch(void* const* d_in, const int* in_sizes, int n_in,
                              void* d_out, int out_size, void* d_ws, size_t ws_size,
                              hipStream_t stream) {
    (void)in_sizes; (void)n_in; (void)out_size; (void)ws_size;
    #define FP(i) ((const float*)d_in[(i)])
    const int* tgt = (const int*)d_in[IN_TGT];
    float* out = (float*)d_out;
    float* out_rot = out + (long)Bb * Ss * Ss;

    // ---- workspace bump allocator ----
    size_t off = 0;
    auto alloc = [&](size_t bytes) -> void* {
        off = (off + 255) & ~(size_t)255;
        void* p = (char*)d_ws + off;
        off += bytes;
        return p;
    };
    // bf16 weights
    unsigned short* w_e0ih = (unsigned short*)alloc((size_t)H4 * CDd * 2);
    unsigned short* w_e0hh = (unsigned short*)alloc((size_t)H4 * Hh * 2);
    unsigned short* w_e1ih = (unsigned short*)alloc((size_t)H4 * Hh * 2);
    unsigned short* w_e1hh = (unsigned short*)alloc((size_t)H4 * Hh * 2);
    unsigned short* w_d0ih = (unsigned short*)alloc((size_t)H4 * DXd * 2);
    unsigned short* w_d0hh = (unsigned short*)alloc((size_t)H4 * Hh * 2);
    unsigned short* w_d1ih = (unsigned short*)alloc((size_t)H4 * Hh * 2);
    unsigned short* w_d1hh = (unsigned short*)alloc((size_t)H4 * Hh * 2);
    unsigned short* w_p1T  = (unsigned short*)alloc((size_t)Hh * DIN * 2);
    unsigned short* w_p2T  = (unsigned short*)alloc((size_t)Hh * Hh * 2);
    unsigned short* w_b2T  = (unsigned short*)alloc((size_t)H2 * H2 * 2);
    unsigned short* w_aqT  = (unsigned short*)alloc((size_t)Hh * Hh * 2);
    unsigned short* w_akT  = (unsigned short*)alloc((size_t)Hh * Hh * 2);
    unsigned short* w_d1T  = (unsigned short*)alloc((size_t)Hh * 2 * Hh * 2);
    unsigned short* w_d2T  = (unsigned short*)alloc((size_t)Ss * Hh * 2);
    unsigned short* w_r1T  = (unsigned short*)alloc((size_t)H2 * 2 * Hh * 2);
    unsigned short* w_r2T  = (unsigned short*)alloc((size_t)4 * H2 * 2);
    // fp32 buffers
    float* bias_e0 = (float*)alloc(H4 * 4);
    float* bias_e1 = (float*)alloc(H4 * 4);
    float* bias_d0 = (float*)alloc(H4 * 4);
    float* bias_d1 = (float*)alloc(H4 * 4);
    float* bin_h1  = (float*)alloc((size_t)Bb * H2 * 4);
    float* bin_enc = (float*)alloc((size_t)Bb * H2 * 4);
    float* comb    = (float*)alloc((size_t)Bb * Ss * CDd * 4);   // combined
    float* X1      = (float*)alloc((size_t)Bb * Ss * Hh * 4);    // parts1 / enc0 out
    float* X2      = (float*)alloc((size_t)Bb * Ss * Hh * 4);    // parts2 / enc (post-ln)
    float* eproj   = (float*)alloc((size_t)Bb * Ss * Hh * 4);    // enc_proj (+att_b)
    float* gX      = (float*)alloc((size_t)Bb * H4 * 4);
    float* gH      = (float*)alloc((size_t)Bb * H4 * 4);
    float* eh      = (float*)alloc((size_t)Bb * Hh * 4);
    float* ec      = (float*)alloc((size_t)Bb * Hh * 4);
    float* dh0     = (float*)alloc((size_t)Bb * Hh * 4);
    float* dc0     = (float*)alloc((size_t)Bb * Hh * 4);
    float* dh1     = (float*)alloc((size_t)Bb * Hh * 4);
    float* dc1     = (float*)alloc((size_t)Bb * Hh * 4);
    float* emean   = (float*)alloc((size_t)Bb * Hh * 4);
    float* dob     = (float*)alloc((size_t)Bb * Hh * 4);
    float* qb      = (float*)alloc((size_t)Bb * Hh * 4);
    float* cc      = (float*)alloc((size_t)Bb * 2 * Hh * 4);
    float* dns1    = (float*)alloc((size_t)Bb * Hh * 4);
    float* lraw    = (float*)alloc((size_t)Bb * Ss * 4);
    float* rot1    = (float*)alloc((size_t)Bb * H2 * 4);
    float* rraw    = (float*)alloc((size_t)Bb * 4 * 4);
    float* dx      = (float*)alloc((size_t)Bb * DXd * 4);
    float* mask    = (float*)alloc((size_t)Bb * Ss * 4);

    // ---- weight conversion (fp32 -> bf16), per call ----
    castw(stream, FP(P_E0_WIH), w_e0ih, (long)H4 * CDd);
    castw(stream, FP(P_E0_WHH), w_e0hh, (long)H4 * Hh);
    castw(stream, FP(P_E1_WIH), w_e1ih, (long)H4 * Hh);
    castw(stream, FP(P_E1_WHH), w_e1hh, (long)H4 * Hh);
    castw(stream, FP(P_D0_WIH), w_d0ih, (long)H4 * DXd);
    castw(stream, FP(P_D0_WHH), w_d0hh, (long)H4 * Hh);
    castw(stream, FP(P_D1_WIH), w_d1ih, (long)H4 * Hh);
    castw(stream, FP(P_D1_WHH), w_d1hh, (long)H4 * Hh);
    castT(stream, FP(P_PART_W1), w_p1T, DIN, Hh);
    castT(stream, FP(P_PART_W2), w_p2T, Hh, Hh);
    castT(stream, FP(P_BIN_W2),  w_b2T, H2, H2);
    castT(stream, FP(P_ATT_W),            w_aqT, Hh, Hh);   // att_w[:H]^T
    castT(stream, FP(P_ATT_W) + Hh * Hh,  w_akT, Hh, Hh);   // att_w[H:]^T
    castT(stream, FP(P_DW1), w_d1T, 2 * Hh, Hh);
    castT(stream, FP(P_DW2), w_d2T, Hh, Ss);
    castT(stream, FP(P_RW1), w_r1T, 2 * Hh, H2);
    castT(stream, FP(P_RW2), w_r2T, H2, 4);
    // folded LSTM biases: bih + bhh
    hipLaunchKernelGGL(k_vadd, dim3(8), dim3(256), 0, stream, FP(P_E0_BIH), FP(P_E0_BHH), bias_e0, (long)H4);
    hipLaunchKernelGGL(k_vadd, dim3(8), dim3(256), 0, stream, FP(P_E1_BIH), FP(P_E1_BHH), bias_e1, (long)H4);
    hipLaunchKernelGGL(k_vadd, dim3(8), dim3(256), 0, stream, FP(P_D0_BIH), FP(P_D0_BHH), bias_d0, (long)H4);
    hipLaunchKernelGGL(k_vadd, dim3(8), dim3(256), 0, stream, FP(P_D1_BIH), FP(P_D1_BHH), bias_d1, (long)H4);

    // ---- bin encoder ----
    hipLaunchKernelGGL(k_bin1, dim3((Bb * H2 + 255) / 256), dim3(256), 0, stream,
                       FP(IN_BIN), FP(P_BIN_W1), FP(P_BIN_B1), bin_h1);
    gemm(stream, bin_h1, H2, w_b2T, FP(P_BIN_B2), bin_enc, H2, Bb, H2, H2, ACT_NONE);

    // ---- parts MLP + combined ----
    gemm(stream, FP(IN_INPUTS), DIN, w_p1T, FP(P_PART_B1), X1, Hh, Bb * Ss, Hh, DIN, ACT_RELU);
    gemm(stream, X1, Hh, w_p2T, FP(P_PART_B2), X2, Hh, Bb * Ss, Hh, Hh, ACT_NONE);
    hipLaunchKernelGGL(k_combined, dim3((unsigned)(((long)Bb * Ss * CDd + 255) / 256)), dim3(256), 0, stream,
                       X2, FP(P_POS), bin_enc, comb);

    // ---- encoder LSTM layer 0: combined -> X1 ----
    zero(stream, eh, (long)Bb * Hh); zero(stream, ec, (long)Bb * Hh);
    for (int t = 0; t < Ss; ++t) {
        gemm(stream, comb + (long)t * CDd, (long)Ss * CDd, w_e0ih, bias_e0, gX, H4, Bb, H4, CDd, ACT_NONE);
        gemm(stream, eh, Hh, w_e0hh, nullptr, gH, H4, Bb, H4, Hh, ACT_NONE);
        hipLaunchKernelGGL(k_lstm_cell, dim3((Bb * Hh + 255) / 256), dim3(256), 0, stream,
                           gX, gH, eh, ec, X1 + (long)t * Hh, (long)Ss * Hh);
    }
    // ---- encoder LSTM layer 1: X1 -> X2 ----
    zero(stream, eh, (long)Bb * Hh); zero(stream, ec, (long)Bb * Hh);
    for (int t = 0; t < Ss; ++t) {
        gemm(stream, X1 + (long)t * Hh, (long)Ss * Hh, w_e1ih, bias_e1, gX, H4, Bb, H4, Hh, ACT_NONE);
        gemm(stream, eh, Hh, w_e1hh, nullptr, gH, H4, Bb, H4, Hh, ACT_NONE);
        hipLaunchKernelGGL(k_lstm_cell, dim3((Bb * Hh + 255) / 256), dim3(256), 0, stream,
                           gX, gH, eh, ec, X2 + (long)t * Hh, (long)Ss * Hh);
    }
    // ---- ln1 (in-place), enc_mean, enc_proj (att_b folded) ----
    hipLaunchKernelGGL(k_layernorm, dim3(Bb * Ss), dim3(256), 0, stream, X2, X2, FP(P_LN1_G), FP(P_LN1_B));
    hipLaunchKernelGGL(k_mean_seq, dim3((Bb * Hh + 255) / 256), dim3(256), 0, stream, X2, emean);
    gemm(stream, X2, Hh, w_akT, FP(P_ATT_B), eproj, Hh, Bb * Ss, Hh, Hh, ACT_NONE);

    // ---- decoder ----
    zero(stream, dh0, (long)Bb * Hh); zero(stream, dc0, (long)Bb * Hh);
    zero(stream, dh1, (long)Bb * Hh); zero(stream, dc1, (long)Bb * Hh);
    zero(stream, mask, (long)Bb * Ss);
    for (int t = 0; t < Ss; ++t) {
        hipLaunchKernelGGL(k_dec_gather, dim3((unsigned)(((long)Bb * DXd + 255) / 256)), dim3(256), 0, stream,
                           comb, emean, tgt, dx, t);
        gemm(stream, dx, DXd, w_d0ih, bias_d0, gX, H4, Bb, H4, DXd, ACT_NONE);
        gemm(stream, dh0, Hh, w_d0hh, nullptr, gH, H4, Bb, H4, Hh, ACT_NONE);
        hipLaunchKernelGGL(k_lstm_cell, dim3((Bb * Hh + 255) / 256), dim3(256), 0, stream,
                           gX, gH, dh0, dc0, (float*)nullptr, 0L);
        gemm(stream, dh0, Hh, w_d1ih, bias_d1, gX, H4, Bb, H4, Hh, ACT_NONE);
        gemm(stream, dh1, Hh, w_d1hh, nullptr, gH, H4, Bb, H4, Hh, ACT_NONE);
        hipLaunchKernelGGL(k_lstm_cell, dim3((Bb * Hh + 255) / 256), dim3(256), 0, stream,
                           gX, gH, dh1, dc1, (float*)nullptr, 0L);
        hipLaunchKernelGGL(k_layernorm, dim3(Bb), dim3(256), 0, stream, dh1, dob, FP(P_LN2_G), FP(P_LN2_B));
        gemm(stream, dob, Hh, w_aqT, nullptr, qb, Hh, Bb, Hh, Hh, ACT_NONE);
        hipLaunchKernelGGL(k_attn, dim3(Bb), dim3(256), 0, stream, qb, eproj, X2, FP(P_ATT_V), dob, cc);
        gemm(stream, cc, 2 * Hh, w_d1T, FP(P_DB1), dns1, Hh, Bb, Hh, 2 * Hh, ACT_RELU);
        gemm(stream, dns1, Hh, w_d2T, FP(P_DB2), lraw, Ss, Bb, Ss, Hh, ACT_NONE);
        hipLaunchKernelGGL(k_logits, dim3(Bb), dim3(256), 0, stream, lraw, mask, tgt, out, t);
        gemm(stream, cc, 2 * Hh, w_r1T, FP(P_RB1), rot1, H2, Bb, H2, 2 * Hh, ACT_RELU);
        gemm(stream, rot1, H2, w_r2T, FP(P_RB2), rraw, 4, Bb, 4, H2, ACT_NONE);
        hipLaunchKernelGGL(k_rot, dim3((Bb * 4 + 255) / 256), dim3(256), 0, stream, rraw, out_rot, t);
    }
    #undef FP
}
